// Encoder_78142634983796
// MI455X (gfx1250) — compile-verified, matrix operand order checked by reading, not run on hardware
//
#include <hip/hip_runtime.h>

// ---------------------------------------------------------------------------
// Transformer encoder forward for MI455X (gfx1250).
//  - bf16 WMMA (v_wmma_f32_16x16x32_bf16) for all GEMMs + flash attention
//  - weights pre-converted to bf16 (halves HBM weight traffic)
//  - Q/K/V kept bf16 in workspace (halves attention staging traffic)
//  - Tensor Data Mover (tensor_load_to_lds + s_wait_tensorcnt) stages Q/K
//    tiles into padded LDS (6-arg builtin variant, per compile feedback)
// B=2, S=2048, D=512, H=8, DH=64, L=6, FFN=2048.
// ---------------------------------------------------------------------------

#define D_MODEL 512
#define D_FF    2048
#define SEQ     2048
#define NB      2
#define NH      8
#define DHEAD   64
#define NLAYER  6
#define ROWS    (NB * SEQ)          // 4096 token rows

typedef __bf16 bf16;
typedef __attribute__((ext_vector_type(16))) __bf16 v16bf;
typedef __attribute__((ext_vector_type(8)))  __bf16 v8bf;
typedef __attribute__((ext_vector_type(8)))  float  v8f;

#if __has_builtin(__builtin_amdgcn_tensor_load_to_lds) && \
    __has_builtin(__builtin_amdgcn_s_wait_tensorcnt)
#define USE_TDM 1
typedef unsigned int u32x4 __attribute__((ext_vector_type(4)));
typedef int          i32x8 __attribute__((ext_vector_type(8)));
typedef int          i32x4 __attribute__((ext_vector_type(4)));
#endif

// Assemble a 16x32 bf16 A/B fragment for one lane from an LDS row.
// ISA layout (16-bit A 16x32): lanes 0-15 hold K=0..7 / K=16..23,
// lanes 16-31 hold K=8..15 / K=24..31; kb = (lane>=16) ? 8 : 0.
__device__ __forceinline__ v16bf load_frag16(const bf16* base) {
    v8bf lo = *(const v8bf*)(base);
    v8bf hi = *(const v8bf*)(base + 16);
    v16bf r;
#pragma unroll
    for (int t = 0; t < 8; ++t) { r[t] = lo[t]; r[8 + t] = hi[t]; }
    return r;
}

__device__ __forceinline__ v8f wmma_bf16(v16bf a, v16bf b, v8f c) {
    return __builtin_amdgcn_wmma_f32_16x16x32_bf16(false, a, false, b,
                                                   (short)0, c, false, false);
}

#ifdef USE_TDM
// Issue one TDM 2-D tile load: bf16 tile (tile_w x tile_h) from a row-major
// tensor with row stride `gstride` elements, into LDS at byte offset
// `lds_off`, padded +4 DWORDs every 32 DWORDs (64 bf16 row -> 72 bf16 LDS
// stride).  Wave-uniform arguments; call from one wave only.
__device__ __forceinline__ void tdm_load_tile_bf16(unsigned lds_off,
                                                   const bf16* gsrc,
                                                   unsigned tile_w,
                                                   unsigned tile_h,
                                                   unsigned long long gstride) {
    unsigned long long ga = (unsigned long long)(uintptr_t)gsrc;
    u32x4 g0;
    g0[0] = 1u;                                         // count=1 (valid D#)
    g0[1] = lds_off;                                    // lds_addr
    g0[2] = (unsigned)(ga & 0xffffffffu);               // global_addr[31:0]
    g0[3] = (unsigned)((ga >> 32) & 0x01ffffffu) | (2u << 30); // [56:32]|type=2
    unsigned long long td0 = gstride;                   // tensor_dim0
    unsigned long long td1 = 1ull << 20;                // tensor_dim1 (large)
    i32x8 g1;
    // data_size=1 (2B), pad_enable, pad_interval=4 (32 DW), pad_amount=3 (4 DW)
    g1[0] = (int)((1u << 16) | (1u << 20) | (4u << 22) | (3u << 25));
    g1[1] = (int)((td0 & 0xffffu) << 16);               // tensor_dim0[15:0]
    g1[2] = (int)(((td0 >> 16) & 0xffffu) | ((td1 & 0xffffu) << 16));
    g1[3] = (int)(((td1 >> 16) & 0xffffu) | (tile_w << 16));
    g1[4] = (int)(tile_h & 0xffffu);                    // tile_dim1 (dim2=0)
    g1[5] = (int)(gstride & 0xffffffffu);               // dim0_stride lo
    g1[6] = (int)((gstride >> 32) & 0xffffu);           // dim0_stride hi
    g1[7] = 0;
    i32x4 gz4 = {0, 0, 0, 0};
    i32x8 gz8 = {0, 0, 0, 0, 0, 0, 0, 0};
    // 6-arg form (amdgpu-toolchain): (g0, g1, g2, g3, g4, cpol)
    __builtin_amdgcn_tensor_load_to_lds(g0, g1, gz4, gz4, gz8, 0);
}
#endif

// ---------------------------------------------------------------------------
// 0) fp32 -> bf16 bulk conversion (weights, run once per launch)
// ---------------------------------------------------------------------------
__global__ void f32_to_bf16_kernel(const float* __restrict__ src,
                                   bf16* __restrict__ dst, int n) {
    int stride = gridDim.x * blockDim.x;
    for (int i = blockIdx.x * blockDim.x + threadIdx.x; i < n; i += stride)
        dst[i] = (bf16)src[i];
}

// ---------------------------------------------------------------------------
// 1) Embedding: h[b,s,:] = word_e[x[b,s],:] + pos_e[s,:]
// ---------------------------------------------------------------------------
__global__ void embed_kernel(const int* __restrict__ x,
                             const float* __restrict__ word_e,
                             const float* __restrict__ pos_e,
                             float* __restrict__ h) {
    int row = blockIdx.x;
    int s   = row & (SEQ - 1);
    int tok = x[row];
    const float* we = word_e + (size_t)tok * D_MODEL;
    const float* pe = pos_e + (size_t)s * D_MODEL;
    float* hr = h + (size_t)row * D_MODEL;
    for (int c = threadIdx.x; c < D_MODEL; c += blockDim.x)
        hr[c] = we[c] + pe[c];
}

// ---------------------------------------------------------------------------
// 2) Tiled GEMM: C = act(A[M,K](f32) * Wb[K,N](bf16) + bias), bf16 WMMA.
//    Block 128x128, K-tile 32, 8 waves each owning a 64x32 patch.
//    Output fp32 (Cf) or bf16 (Cb) -- exactly one is non-null.
// ---------------------------------------------------------------------------
#define LDT 40   // bf16 LDS stride (80B, 16B multiple)

__global__ void gemm_kernel(const float* __restrict__ A,
                            const bf16* __restrict__ Wb,
                            const float* __restrict__ bias,
                            float* __restrict__ Cf,
                            bf16* __restrict__ Cb,
                            int M, int N, int Kd, int relu) {
    __shared__ __align__(16) bf16 a_s[128 * LDT];
    __shared__ __align__(16) bf16 b_s[128 * LDT];   // stored [n][k]

    const int bm   = blockIdx.x;
    const int bn   = blockIdx.y;
    const int tid  = threadIdx.x;
    const int wave = tid >> 5;
    const int lane = tid & 31;
    const int wm   = wave >> 2;
    const int wn   = wave & 3;
    const int lr   = lane & 15;
    const int kb   = (lane >> 4) << 3;

    v8f acc[4][2] = {};

    const int ar  = tid >> 1;            // A loader: row 0..127
    const int ac0 = (tid & 1) * 16;
    const int wr  = tid >> 3;            // W loader: k row 0..31
    const int wc0 = (tid & 7) * 16;

    for (int k0 = 0; k0 < Kd; k0 += 32) {
        {   // A tile: fp32 -> bf16
            const float* src = A + (size_t)(bm * 128 + ar) * Kd + k0 + ac0;
#pragma unroll
            for (int j = 0; j < 16; ++j) a_s[ar * LDT + ac0 + j] = (bf16)src[j];
        }
        {   // W tile: bf16 copy, transposed store to [n][k]
            const bf16* src = Wb + (size_t)(k0 + wr) * N + bn * 128 + wc0;
            if (k0 + 32 < Kd)
                __builtin_prefetch(Wb + (size_t)(k0 + 32 + wr) * N + bn * 128 + wc0, 0, 1);
            v8bf w0 = *(const v8bf*)src;
            v8bf w1 = *(const v8bf*)(src + 8);
#pragma unroll
            for (int j = 0; j < 8; ++j) {
                b_s[(wc0 + j) * LDT + wr]     = w0[j];
                b_s[(wc0 + 8 + j) * LDT + wr] = w1[j];
            }
        }
        __syncthreads();

        v16bf af[4], bfr[2];
#pragma unroll
        for (int ti = 0; ti < 4; ++ti)
            af[ti] = load_frag16(&a_s[(wm * 64 + ti * 16 + lr) * LDT + kb]);
#pragma unroll
        for (int tj = 0; tj < 2; ++tj)
            bfr[tj] = load_frag16(&b_s[(wn * 32 + tj * 16 + lr) * LDT + kb]);
#pragma unroll
        for (int ti = 0; ti < 4; ++ti)
#pragma unroll
            for (int tj = 0; tj < 2; ++tj)
                acc[ti][tj] = wmma_bf16(af[ti], bfr[tj], acc[ti][tj]);
        __syncthreads();
    }

    const int rhalf = (lane >> 4) << 3;
#pragma unroll
    for (int ti = 0; ti < 4; ++ti) {
#pragma unroll
        for (int tj = 0; tj < 2; ++tj) {
            int col = bn * 128 + wn * 32 + tj * 16 + lr;
            float bv = bias ? bias[col] : 0.0f;
#pragma unroll
            for (int vi = 0; vi < 8; ++vi) {
                int row = bm * 128 + wm * 64 + ti * 16 + rhalf + vi;
                float v = acc[ti][tj][vi] + bv;
                if (relu) v = fmaxf(v, 0.0f);
                if (Cb) Cb[(size_t)row * N + col] = (bf16)v;
                else    Cf[(size_t)row * N + col] = v;
            }
        }
    }
}

// ---------------------------------------------------------------------------
// 3) Flash attention: one workgroup per (b, head, 64-row q block).
//    Q/K/V are bf16.  Q and K tiles staged by the Tensor Data Mover (with
//    D# LDS padding giving stride 72); V staged transposed by the workgroup.
//    Online softmax over 32 KV tiles of 64.
// ---------------------------------------------------------------------------
#define LDA 72   // bf16 LDS stride (144B, 16B multiple)

__global__ void flash_attn_kernel(const bf16* __restrict__ Qb,
                                  const bf16* __restrict__ Kb,
                                  const bf16* __restrict__ Vb,
                                  const int* __restrict__ mask,
                                  float* __restrict__ O) {
    __shared__ __align__(16) bf16 q_s[64 * LDA];   // [i][d]
    __shared__ __align__(16) bf16 k_s[64 * LDA];   // [j][d]
    __shared__ __align__(16) bf16 vT_s[64 * LDA];  // [d][j]
    __shared__ __align__(16) bf16 p_s[64 * LDA];   // [i][j]
    __shared__ float s_f[64 * 64];
    __shared__ float m_s[64], l_s[64], f_s[64];

    const int bid = blockIdx.x;
    const int qb  = bid & 31;
    const int hh  = (bid >> 5) & (NH - 1);
    const int b   = bid >> 8;
    const int tid = threadIdx.x;
    const int lane = tid & 31;
    const int ti   = (tid >> 5) >> 1;     // i-tile 0..3
    const int tdb  = ((tid >> 5) & 1) * 2;
    const int lr   = lane & 15;
    const int kb   = (lane >> 4) << 3;
    const int rhalf = (lane >> 4) << 3;
    const float scale = 0.04419417382415922f;   // 1/sqrt(512)

    const size_t qbase = ((size_t)(b * SEQ + qb * 64)) * D_MODEL + hh * DHEAD;

    // ---- stage Q tile ----
#ifdef USE_TDM
    if (tid < 32)
        tdm_load_tile_bf16((unsigned)(uintptr_t)q_s, Qb + qbase, 64, 64, D_MODEL);
#else
    for (int e = tid; e < 64 * 8; e += 256) {
        int i = e >> 3, dc = (e & 7) * 8;
        *(v8bf*)&q_s[i * LDA + dc] =
            *(const v8bf*)&Qb[qbase + (size_t)i * D_MODEL + dc];
    }
#endif
    if (tid < 64) { m_s[tid] = -3.0e38f; l_s[tid] = 0.0f; }

    v8f acc[2] = {};
    for (int kt = 0; kt < SEQ / 64; ++kt) {
        const size_t kvbase = ((size_t)(b * SEQ + kt * 64)) * D_MODEL + hh * DHEAD;
        // ---- stage K (TDM) and V transposed (manual, overlapped) ----
#ifdef USE_TDM
        if (tid < 32)
            tdm_load_tile_bf16((unsigned)(uintptr_t)k_s, Kb + kvbase, 64, 64, D_MODEL);
#else
        for (int e = tid; e < 64 * 8; e += 256) {
            int j = e >> 3, dc = (e & 7) * 8;
            *(v8bf*)&k_s[j * LDA + dc] =
                *(const v8bf*)&Kb[kvbase + (size_t)j * D_MODEL + dc];
        }
#endif
        for (int e = tid; e < 64 * 64; e += 256) {
            int j = e >> 6, d = e & 63;
            vT_s[d * LDA + j] = Vb[kvbase + (size_t)j * D_MODEL + d];
        }
#ifdef USE_TDM
        if (tid < 32) __builtin_amdgcn_s_wait_tensorcnt(0);
#endif
        __syncthreads();

        // ---- S = Q K^T (two 16x16 score tiles per wave) ----
        v16bf aq0 = load_frag16(&q_s[(ti * 16 + lr) * LDA + kb]);
        v16bf aq1 = load_frag16(&q_s[(ti * 16 + lr) * LDA + kb + 32]);
#pragma unroll
        for (int t = 0; t < 2; ++t) {
            int tj = tdb + t;
            v8f sa = {};
            v16bf bk0 = load_frag16(&k_s[(tj * 16 + lr) * LDA + kb]);
            v16bf bk1 = load_frag16(&k_s[(tj * 16 + lr) * LDA + kb + 32]);
            sa = wmma_bf16(aq0, bk0, sa);
            sa = wmma_bf16(aq1, bk1, sa);
            int jg = kt * 64 + tj * 16 + lr;
            bool masked = (mask[b * SEQ + jg] == 0);
#pragma unroll
            for (int vi = 0; vi < 8; ++vi) {
                int il = ti * 16 + rhalf + vi;
                s_f[il * 64 + tj * 16 + lr] = masked ? -1.0e20f : sa[vi] * scale;
            }
        }
        __syncthreads();

        // ---- online softmax (one row per thread, threads 0..63) ----
        if (tid < 64) {
            int r = tid;
            float mold = m_s[r], mnew = mold;
#pragma unroll 8
            for (int j = 0; j < 64; ++j) mnew = fmaxf(mnew, s_f[r * 64 + j]);
            float fac = __expf(mold - mnew);
            float ls  = l_s[r] * fac;
#pragma unroll 8
            for (int j = 0; j < 64; ++j) {
                float p = __expf(s_f[r * 64 + j] - mnew);
                p_s[r * LDA + j] = (bf16)p;
                ls += p;
            }
            m_s[r] = mnew; l_s[r] = ls; f_s[r] = fac;
        }
        __syncthreads();

        // ---- rescale running context, ctx += P V ----
        float facv[8];
#pragma unroll
        for (int vi = 0; vi < 8; ++vi) facv[vi] = f_s[ti * 16 + rhalf + vi];
        v16bf ap0 = load_frag16(&p_s[(ti * 16 + lr) * LDA + kb]);
        v16bf ap1 = load_frag16(&p_s[(ti * 16 + lr) * LDA + kb + 32]);
#pragma unroll
        for (int t = 0; t < 2; ++t) {
#pragma unroll
            for (int vi = 0; vi < 8; ++vi) acc[t][vi] *= facv[vi];
            int td = tdb + t;
            v16bf bv0 = load_frag16(&vT_s[(td * 16 + lr) * LDA + kb]);
            v16bf bv1 = load_frag16(&vT_s[(td * 16 + lr) * LDA + kb + 32]);
            acc[t] = wmma_bf16(ap0, bv0, acc[t]);
            acc[t] = wmma_bf16(ap1, bv1, acc[t]);
        }
        __syncthreads();
    }

    // ---- epilogue: O = ctx / l (fp32) ----
#pragma unroll
    for (int t = 0; t < 2; ++t) {
        int dl = (tdb + t) * 16 + lr;
#pragma unroll
        for (int vi = 0; vi < 8; ++vi) {
            int il = ti * 16 + rhalf + vi;
            float o = acc[t][vi] / l_s[il];
            O[((size_t)(b * SEQ + qb * 64 + il)) * D_MODEL + hh * DHEAD + dl] = o;
        }
    }
}

// ---------------------------------------------------------------------------
// 4) Fused residual-add + LayerNorm over D=512, one block (256 thr) per row.
// ---------------------------------------------------------------------------
__global__ void add_ln_kernel(const float* __restrict__ A,
                              const float* __restrict__ Bv,
                              const float* __restrict__ gamma,
                              const float* __restrict__ beta,
                              float* __restrict__ out) {
    __shared__ float r1[256];
    __shared__ float r2[256];
    const int row = blockIdx.x;
    const int tid = threadIdx.x;
    const size_t base = (size_t)row * D_MODEL;

    float x0 = A[base + tid]       + Bv[base + tid];
    float x1 = A[base + 256 + tid] + Bv[base + 256 + tid];
    r1[tid] = x0 + x1;
    r2[tid] = x0 * x0 + x1 * x1;
    __syncthreads();
#pragma unroll
    for (int off = 128; off > 0; off >>= 1) {
        if (tid < off) { r1[tid] += r1[tid + off]; r2[tid] += r2[tid + off]; }
        __syncthreads();
    }
    float mean = r1[0] * (1.0f / D_MODEL);
    float var  = r2[0] * (1.0f / D_MODEL) - mean * mean;
    float rstd = rsqrtf(var + 1e-5f);
    out[base + tid]       = (x0 - mean) * rstd * gamma[tid]       + beta[tid];
    out[base + 256 + tid] = (x1 - mean) * rstd * gamma[256 + tid] + beta[256 + tid];
}

// ---------------------------------------------------------------------------
// Orchestration
// ---------------------------------------------------------------------------
extern "C" void kernel_launch(void* const* d_in, const int* in_sizes, int n_in,
                              void* d_out, int out_size, void* d_ws, size_t ws_size,
                              hipStream_t stream) {
    const int*   x      = (const int*)  d_in[0];
    const int*   mask   = (const int*)  d_in[1];
    const float* word_e = (const float*)d_in[2];
    const float* pos_e  = (const float*)d_in[3];
    const float* Wv     = (const float*)d_in[4];
    const float* Wk     = (const float*)d_in[5];
    const float* Wq     = (const float*)d_in[6];
    const float* Wo     = (const float*)d_in[7];
    const float* bo     = (const float*)d_in[8];
    const float* W1     = (const float*)d_in[9];
    const float* b1     = (const float*)d_in[10];
    const float* W2     = (const float*)d_in[11];
    const float* b2     = (const float*)d_in[12];
    const float* gamma  = (const float*)d_in[13];
    const float* beta   = (const float*)d_in[14];
    float* out = (float*)d_out;

    const size_t T      = (size_t)ROWS * D_MODEL;          // 2M elements
    const int    DD_ALL = NLAYER * D_MODEL * D_MODEL;      // per proj weights
    const int    DF_ALL = NLAYER * D_MODEL * D_FF;         // per FFN weights

    char* wsp = (char*)d_ws;
    auto carve = [&wsp](size_t bytes) {
        char* p = wsp;
        wsp += (bytes + 255) & ~(size_t)255;
        return p;
    };
    float* h    = (float*)carve(T * 4);
    float* h1   = (float*)carve(T * 4);
    float* tmp  = (float*)carve(T * 4);
    float* ctx  = (float*)carve(T * 4);
    float* ff1  = (float*)carve((size_t)ROWS * D_FF * 4);
    bf16*  qbf  = (bf16*)carve(T * 2);
    bf16*  kbf  = (bf16*)carve(T * 2);
    bf16*  vbf  = (bf16*)carve(T * 2);
    bf16*  Wq_b = (bf16*)carve((size_t)DD_ALL * 2);
    bf16*  Wk_b = (bf16*)carve((size_t)DD_ALL * 2);
    bf16*  Wv_b = (bf16*)carve((size_t)DD_ALL * 2);
    bf16*  Wo_b = (bf16*)carve((size_t)DD_ALL * 2);
    bf16*  W1_b = (bf16*)carve((size_t)DF_ALL * 2);
    bf16*  W2_b = (bf16*)carve((size_t)DF_ALL * 2);

    const dim3 blk(256);
    const dim3 gProj(ROWS / 128, D_MODEL / 128);      // 32 x 4
    const dim3 gFF1 (ROWS / 128, D_FF / 128);         // 32 x 16
    const dim3 gAttn(NB * NH * (SEQ / 64));           // 512

    // one-time (per launch) weight conversion to bf16
    f32_to_bf16_kernel<<<2048, blk, 0, stream>>>(Wq, Wq_b, DD_ALL);
    f32_to_bf16_kernel<<<2048, blk, 0, stream>>>(Wk, Wk_b, DD_ALL);
    f32_to_bf16_kernel<<<2048, blk, 0, stream>>>(Wv, Wv_b, DD_ALL);
    f32_to_bf16_kernel<<<2048, blk, 0, stream>>>(Wo, Wo_b, DD_ALL);
    f32_to_bf16_kernel<<<4096, blk, 0, stream>>>(W1, W1_b, DF_ALL);
    f32_to_bf16_kernel<<<4096, blk, 0, stream>>>(W2, W2_b, DF_ALL);

    embed_kernel<<<ROWS, blk, 0, stream>>>(x, word_e, pos_e, h);

    for (int l = 0; l < NLAYER; ++l) {
        const size_t oDD = (size_t)l * D_MODEL * D_MODEL;
        const size_t oDF = (size_t)l * D_MODEL * D_FF;
        const size_t oD  = (size_t)l * D_MODEL;
        const size_t oF  = (size_t)l * D_FF;

        // Q/K/V projections (bf16 outputs)
        gemm_kernel<<<gProj, blk, 0, stream>>>(h, Wq_b + oDD, nullptr,
                                               nullptr, qbf, ROWS, D_MODEL, D_MODEL, 0);
        gemm_kernel<<<gProj, blk, 0, stream>>>(h, Wk_b + oDD, nullptr,
                                               nullptr, kbf, ROWS, D_MODEL, D_MODEL, 0);
        gemm_kernel<<<gProj, blk, 0, stream>>>(h, Wv_b + oDD, nullptr,
                                               nullptr, vbf, ROWS, D_MODEL, D_MODEL, 0);
        // Attention (fp32 context out)
        flash_attn_kernel<<<gAttn, blk, 0, stream>>>(qbf, kbf, vbf, mask, ctx);
        // Output projection + bias
        gemm_kernel<<<gProj, blk, 0, stream>>>(ctx, Wo_b + oDD, bo + oD,
                                               tmp, nullptr, ROWS, D_MODEL, D_MODEL, 0);
        // h1 = LN(attn_out + h)
        add_ln_kernel<<<ROWS, blk, 0, stream>>>(tmp, h, gamma + oD, beta + oD, h1);
        // FFN
        gemm_kernel<<<gFF1, blk, 0, stream>>>(h1, W1_b + oDF, b1 + oF,
                                              ff1, nullptr, ROWS, D_FF, D_MODEL, 1);
        gemm_kernel<<<gProj, blk, 0, stream>>>(ff1, W2_b + oDF, b2 + oD,
                                               tmp, nullptr, ROWS, D_MODEL, D_FF, 0);
        // h = LN(h1 + ff)  (last layer writes straight to d_out)
        float* dst = (l == NLAYER - 1) ? out : h;
        add_ln_kernel<<<ROWS, blk, 0, stream>>>(h1, tmp, gamma + oD, beta + oD, dst);
    }
}
